// SimpleLLM_69578470195640
// MI455X (gfx1250) — compile-verified, matrix operand order checked by reading, not run on hardware
//
#include <hip/hip_runtime.h>
#include <stdint.h>

// ---------------------------------------------------------------------------
// SimpleLLM forward for MI455X (gfx1250, wave32, WMMA + TDM).
//  - GEMMs: bf16 activations x fp32->bf16 weights, f32 accumulate via
//    v_wmma_f32_16x16x32_bf16; 64x64 tile per wave, 256x128 per block.
//    Weight tiles staged into LDS by the Tensor Data Mover
//    (tensor_load_to_lds, statically double-buffered, TENSORcnt-tracked,
//    TDM LDS padding for bank-conflict-free reads).
//  - Attention: WMMA flash attention, online softmax, LDS relayout of P.
// ---------------------------------------------------------------------------

typedef __bf16 bf16_t;
typedef float   v4f   __attribute__((ext_vector_type(4)));
typedef float   v8f   __attribute__((ext_vector_type(8)));
typedef bf16_t  v8bf  __attribute__((ext_vector_type(8)));
typedef bf16_t  v16bf __attribute__((ext_vector_type(16)));
typedef int     v4i   __attribute__((ext_vector_type(4)));
typedef int     v8i   __attribute__((ext_vector_type(8)));

#define C_H   2048
#define C_NH  32
#define C_HD  64
#define C_KV  8
#define C_GS  4
#define C_L   4
#define C_F   8192
#define C_V   32000
#define C_S   2048

// LDS staging: 128 rows x 32 f32, padded by TDM to 36-dword row stride
#define BSTRIDE 36
#define BTILE_BYTES (128 * BSTRIDE * 4)

// ---------------------------------------------------------------------------
// TDM: issue a 2D tile load (128 rows x 32 f32) of B[N,K] into LDS.
//   D# group0 (ISA 8.3): count=1, lds_addr, 57b global_addr, type=2
//   D# group1 (ISA 8.4): data_size=4B(2), pad_enable, pad_interval=4 (32 dw),
//                        pad_amount=3 (4 dw), tensor dims, tile 32x128, stride K
// The LDS destination pointer is passed as an (unused) asm input so the
// staging buffer ESCAPES: alias analysis must then assume the asm writes it,
// keeping the ds_load fragment reads alive.
// ---------------------------------------------------------------------------
static __device__ __forceinline__
void tdm_load_b_tile(const float* src, int K, int N,
                     float* lds_dst, unsigned lds_off) {
  uint64_t ga = (uint64_t)(uintptr_t)src;
  v4i g0;
  g0[0] = 1;                                             // count=1, is_restore=0
  g0[1] = (int)lds_off;                                  // lds_addr
  g0[2] = (int)(uint32_t)(ga & 0xFFFFFFFFu);             // global_addr[31:0]
  g0[3] = (int)((uint32_t)((ga >> 32) & 0x1FFFFFFu)      // global_addr[56:32]
                | (2u << 30));                           // type = 2 ("image")
  uint32_t tdim0 = (uint32_t)K;                          // tensor len dim0
  uint32_t tdim1 = (uint32_t)N;                          // tensor len dim1
  uint64_t st0   = (uint64_t)(uint32_t)K;                // dim0 stride (elems)
  v8i g1;
  // d0: wg_mask=0 | data_size=2(4B)<<16 | pad_en<<20 | pad_interval=4<<22
  //     | pad_amount=3<<25   (rows land on 36-dword stride in LDS)
  g1[0] = (int)(0x20000u | (1u << 20) | (4u << 22) | (3u << 25));
  g1[1] = (int)((tdim0 & 0xFFFFu) << 16);                // tensor_dim0[15:0]
  g1[2] = (int)(((tdim0 >> 16) & 0xFFFFu) | ((tdim1 & 0xFFFFu) << 16));
  g1[3] = (int)(((tdim1 >> 16) & 0xFFFFu) | (32u << 16));// tile_dim0 = 32
  g1[4] = 128;                                           // tile_dim1 = 128
  g1[5] = (int)(uint32_t)(st0 & 0xFFFFFFFFu);            // stride0[31:0]
  g1[6] = (int)((uint32_t)(st0 >> 32) & 0xFFFFu);        // stride0[47:32]
  g1[7] = 0;
  asm volatile("tensor_load_to_lds %0, %1"
               :: "s"(g0), "s"(g1), "s"(lds_dst)
               : "memory");
}

// ---------------------------------------------------------------------------
// Embedding gather: h[s, :] = embed[ids[s], :]
// ---------------------------------------------------------------------------
__global__ void embed_gather_kernel(const int* __restrict__ ids,
                                    const float* __restrict__ embed,
                                    float* __restrict__ h) {
  int idx = blockIdx.x * 256 + threadIdx.x;
  if (idx >= C_S * C_H) return;
  int s = idx >> 11;
  int j = idx & (C_H - 1);
  h[idx] = embed[(size_t)ids[s] * C_H + j];
}

// ---------------------------------------------------------------------------
// RMSNorm -> bf16 activations.  One 256-thread block per row (H = 2048)
// ---------------------------------------------------------------------------
__global__ void rmsnorm_bf16_kernel(const float* __restrict__ x,
                                    const float* __restrict__ w,
                                    bf16_t* __restrict__ out) {
  int row = blockIdx.x;
  const float* xr = x + (size_t)row * C_H;
  float vals[8];
  float ss = 0.f;
#pragma unroll
  for (int i = 0; i < 8; i++) {
    float v = xr[threadIdx.x + i * 256];
    vals[i] = v;
    ss += v * v;
  }
  for (int off = 16; off >= 1; off >>= 1) ss += __shfl_xor(ss, off, 32);
  __shared__ float red[8];
  int wave = threadIdx.x >> 5, lane = threadIdx.x & 31;
  if (lane == 0) red[wave] = ss;
  __syncthreads();
  float tot = 0.f;
#pragma unroll
  for (int wv = 0; wv < 8; wv++) tot += red[wv];
  float r = rsqrtf(tot * (1.0f / (float)C_H) + 1e-5f);
#pragma unroll
  for (int i = 0; i < 8; i++) {
    int j = threadIdx.x + i * 256;
    out[(size_t)row * C_H + j] = (bf16_t)(vals[i] * r * w[j]);
  }
}

// ---------------------------------------------------------------------------
// One K-step (32) of the WMMA GEMM: read B tile from LDS, A from global.
// ---------------------------------------------------------------------------
static __device__ __forceinline__
void gemm_kstep(const bf16_t* a0, const float* btile, int wn, int l16, int hi,
                int k0, int K, v8f* acc) {
  v16bf bfrag[4];
#pragma unroll
  for (int jn = 0; jn < 4; jn++) {
    const float* bp = btile + (wn * 64 + jn * 16 + l16) * BSTRIDE + hi * 8;
    v4f f0 = *(const v4f*)(bp);
    v4f f1 = *(const v4f*)(bp + 4);
    v4f f2 = *(const v4f*)(bp + 16);
    v4f f3 = *(const v4f*)(bp + 20);
#pragma unroll
    for (int e = 0; e < 4; e++) {
      bfrag[jn][e]      = (bf16_t)f0[e];
      bfrag[jn][4 + e]  = (bf16_t)f1[e];
      bfrag[jn][8 + e]  = (bf16_t)f2[e];
      bfrag[jn][12 + e] = (bf16_t)f3[e];
    }
  }
#pragma unroll
  for (int im = 0; im < 4; im++) {
    const bf16_t* ap = a0 + (size_t)im * 16 * K + k0;
    v8bf alo = *(const v8bf*)(ap);
    v8bf ahi = *(const v8bf*)(ap + 16);
    v16bf afrag;
#pragma unroll
    for (int e = 0; e < 8; e++) {
      afrag[e]     = alo[e];
      afrag[8 + e] = ahi[e];
    }
#pragma unroll
    for (int jn = 0; jn < 4; jn++) {
      acc[im * 4 + jn] = __builtin_amdgcn_wmma_f32_16x16x32_bf16(
          false, afrag, false, bfrag[jn], (short)0, acc[im * 4 + jn],
          false, false);
    }
  }
}

// ---------------------------------------------------------------------------
// WMMA GEMM with TDM-staged weights:
//   C[M,N] (+)= A[M,K](bf16) * B[N,K](f32->bf16)^T
// Block: 8 waves = 4(M) x 2(N), wave tile 64x64, block tile 256x128.
// K unrolled by 64: static double-buffer (phase0: compute buf0 / TDM->buf1,
// phase1: compute buf1 / TDM->buf0).  Requires K % 64 == 0 (true here).
// Grid: (N/128, M/256).
// ---------------------------------------------------------------------------
__global__ __launch_bounds__(256)
void gemm_bf16_wmma_kernel(const bf16_t* __restrict__ A,
                           const float* __restrict__ B,
                           float* __restrict__ C,
                           int M, int N, int K, int accumulate) {
  __shared__ float bstage[2][128 * BSTRIDE];

  const int lane = threadIdx.x & 31;
  const int wave = threadIdx.x >> 5;
  const int l16  = lane & 15;
  const int hi   = lane >> 4;
  const int wm   = wave >> 1;          // 0..3
  const int wn   = wave & 1;           // 0..1
  const int m0   = blockIdx.y * 256 + wm * 64;
  const int n0b  = blockIdx.x * 128;
  const int n0   = n0b + wn * 64;

  v8f zero = {};
  v8f acc[16];
#pragma unroll
  for (int i = 0; i < 16; i++) acc[i] = zero;

  const bf16_t* a0   = A + (size_t)(m0 + l16) * K + hi * 8;
  const float*  bsrc = B + (size_t)n0b * K;

  if (wave == 0) tdm_load_b_tile(bsrc, K, N, &bstage[0][0], 0);

  for (int k0 = 0; k0 < K; k0 += 64) {
    // ---- phase 0: compute from buf0, TDM fills buf1 with k0+32 ----
    if (wave == 0) __builtin_amdgcn_s_wait_tensorcnt(0);
    __syncthreads();
    if (wave == 0)
      tdm_load_b_tile(bsrc + k0 + 32, K, N, &bstage[1][0], BTILE_BYTES);
    gemm_kstep(a0, &bstage[0][0], wn, l16, hi, k0, K, acc);
    __syncthreads();
    // ---- phase 1: compute from buf1, TDM fills buf0 with k0+64 ----
    if (wave == 0) __builtin_amdgcn_s_wait_tensorcnt(0);
    __syncthreads();
    if (wave == 0 && (k0 + 64) < K)
      tdm_load_b_tile(bsrc + k0 + 64, K, N, &bstage[0][0], 0);
    gemm_kstep(a0, &bstage[1][0], wn, l16, hi, k0 + 32, K, acc);
    __syncthreads();
  }

#pragma unroll
  for (int im = 0; im < 4; im++) {
#pragma unroll
    for (int jn = 0; jn < 4; jn++) {
      float* cp = C + (size_t)(m0 + im * 16 + hi * 8) * N + n0 + jn * 16 + l16;
      v8f cv = acc[im * 4 + jn];
#pragma unroll
      for (int i = 0; i < 8; i++) {
        float* dst = cp + (size_t)i * N;
        if (accumulate) *dst += cv[i];
        else            *dst  = cv[i];
      }
    }
  }
}

// ---------------------------------------------------------------------------
// RoPE: read fp32 [S, nheads, 64], write rotated bf16 (same layout)
// ---------------------------------------------------------------------------
__global__ void rope_bf16_kernel(const float* __restrict__ x,
                                 bf16_t* __restrict__ out,
                                 int nheads, int total) {
  int idx = blockIdx.x * 256 + threadIdx.x;
  if (idx >= total) return;
  int i = idx & 31;
  int t = idx >> 5;
  int head = t % nheads;
  int s = t / nheads;
  float inv = __expf(-13.1223633774f * ((float)i * (1.0f / 32.0f))); // 500000^(-i/32)
  float ang = (float)s * inv;
  float c = __cosf(ang), sn = __sinf(ang);
  size_t off = ((size_t)s * nheads + head) * 64;
  float xe = x[off + i], xo = x[off + i + 32];
  out[off + i]      = (bf16_t)(xe * c - xo * sn);
  out[off + i + 32] = (bf16_t)(xo * c + xe * sn);
}

// ---------------------------------------------------------------------------
// V transpose+convert: v[S, KV, HD] f32 -> vt[KV*HD, S] bf16
// ---------------------------------------------------------------------------
__global__ void v_transpose_bf16_kernel(const float* __restrict__ v,
                                        bf16_t* __restrict__ vt) {
  int idx = blockIdx.x * 256 + threadIdx.x;
  if (idx >= C_S * C_KV * C_HD) return;
  int s = idx >> 9;
  int r = idx & 511;
  vt[(size_t)r * C_S + s] = (bf16_t)v[idx];
}

// ---------------------------------------------------------------------------
// WMMA flash attention (known-good codegen).
// ---------------------------------------------------------------------------
__global__ __launch_bounds__(256)
void flash_attn_wmma_kernel(const bf16_t* __restrict__ qb,
                            const bf16_t* __restrict__ kb,
                            const bf16_t* __restrict__ vt,
                            bf16_t* __restrict__ out) {
  const int lane = threadIdx.x & 31;
  const int wave = threadIdx.x >> 5;
  const int l16  = lane & 15;
  const int hi   = lane >> 4;
  const int head = blockIdx.y;
  const int kvh  = head >> 2;
  const int q0   = blockIdx.x * 128 + wave * 16;

  __shared__ bf16_t pbuf[8][16 * 32];

  v16bf qf[2];
  {
    const bf16_t* qrow =
        qb + ((size_t)(q0 + l16) * C_NH + head) * C_HD + hi * 8;
#pragma unroll
    for (int ks = 0; ks < 2; ks++) {
      v8bf lo = *(const v8bf*)(qrow + ks * 32);
      v8bf hh = *(const v8bf*)(qrow + ks * 32 + 16);
#pragma unroll
      for (int e = 0; e < 8; e++) { qf[ks][e] = lo[e]; qf[ks][8 + e] = hh[e]; }
    }
  }

  v8f zero = {};
  v8f oacc[4];
#pragma unroll
  for (int f = 0; f < 4; f++) oacc[f] = zero;
  v8f mrow, drow;
#pragma unroll
  for (int e = 0; e < 8; e++) { mrow[e] = -1e30f; drow[e] = 0.f; }

  const float scale = 0.125f;
  const int tmax = q0 + 15;

  for (int t0 = 0; t0 <= tmax; t0 += 32) {
    v8f sacc[2];
    sacc[0] = zero; sacc[1] = zero;
#pragma unroll
    for (int ks = 0; ks < 2; ks++) {
#pragma unroll
      for (int jn = 0; jn < 2; jn++) {
        const bf16_t* kp =
            kb + ((size_t)(t0 + jn * 16 + l16) * C_KV + kvh) * C_HD +
            ks * 32 + hi * 8;
        v8bf lo = *(const v8bf*)kp;
        v8bf hh = *(const v8bf*)(kp + 16);
        v16bf bf_;
#pragma unroll
        for (int e = 0; e < 8; e++) { bf_[e] = lo[e]; bf_[8 + e] = hh[e]; }
        sacc[jn] = __builtin_amdgcn_wmma_f32_16x16x32_bf16(
            false, qf[ks], false, bf_, (short)0, sacc[jn], false, false);
      }
    }

    const bool boundary = (t0 + 31 > q0);
#pragma unroll
    for (int jn = 0; jn < 2; jn++) {
#pragma unroll
      for (int e = 0; e < 8; e++) {
        float s = sacc[jn][e] * scale;
        if (boundary) {
          int key = t0 + jn * 16 + l16;
          int qq  = q0 + e + hi * 8;
          s = (key > qq) ? -1e30f : s;
        }
        sacc[jn][e] = s;
      }
    }

    v8f tmx;
#pragma unroll
    for (int e = 0; e < 8; e++) tmx[e] = fmaxf(sacc[0][e], sacc[1][e]);
    for (int msk = 1; msk <= 8; msk <<= 1)
#pragma unroll
      for (int e = 0; e < 8; e++)
        tmx[e] = fmaxf(tmx[e], __shfl_xor(tmx[e], msk, 32));

    v8f mnew, corr;
#pragma unroll
    for (int e = 0; e < 8; e++) {
      mnew[e] = fmaxf(mrow[e], tmx[e]);
      corr[e] = __expf(mrow[e] - mnew[e]);
      mrow[e] = mnew[e];
    }
#pragma unroll
    for (int jn = 0; jn < 2; jn++)
#pragma unroll
      for (int e = 0; e < 8; e++)
        sacc[jn][e] = __expf(sacc[jn][e] - mnew[e]);

    v8f tsm;
#pragma unroll
    for (int e = 0; e < 8; e++) tsm[e] = sacc[0][e] + sacc[1][e];
    for (int msk = 1; msk <= 8; msk <<= 1)
#pragma unroll
      for (int e = 0; e < 8; e++) tsm[e] += __shfl_xor(tsm[e], msk, 32);
#pragma unroll
    for (int e = 0; e < 8; e++) drow[e] = drow[e] * corr[e] + tsm[e];
#pragma unroll
    for (int f = 0; f < 4; f++)
#pragma unroll
      for (int e = 0; e < 8; e++) oacc[f][e] *= corr[e];

    bf16_t* pw = &pbuf[wave][0];
#pragma unroll
    for (int jn = 0; jn < 2; jn++)
#pragma unroll
      for (int e = 0; e < 8; e++)
        pw[(e + hi * 8) * 32 + jn * 16 + l16] = (bf16_t)sacc[jn][e];
    asm volatile("s_wait_dscnt 0x0" ::: "memory");
    v16bf pf;
    {
      const bf16_t* pr = pw + l16 * 32 + hi * 8;
      v8bf lo = *(const v8bf*)(pr);
      v8bf hh = *(const v8bf*)(pr + 16);
#pragma unroll
      for (int e = 0; e < 8; e++) { pf[e] = lo[e]; pf[8 + e] = hh[e]; }
    }

#pragma unroll
    for (int jn = 0; jn < 4; jn++) {
      const bf16_t* vp =
          vt + (size_t)(kvh * 64 + jn * 16 + l16) * C_S + t0 + hi * 8;
      v8bf lo = *(const v8bf*)vp;
      v8bf hh = *(const v8bf*)(vp + 16);
      v16bf bv;
#pragma unroll
      for (int e = 0; e < 8; e++) { bv[e] = lo[e]; bv[8 + e] = hh[e]; }
      oacc[jn] = __builtin_amdgcn_wmma_f32_16x16x32_bf16(
          false, pf, false, bv, (short)0, oacc[jn], false, false);
    }
  }

  v8f inv;
#pragma unroll
  for (int e = 0; e < 8; e++) inv[e] = 1.0f / drow[e];
#pragma unroll
  for (int jn = 0; jn < 4; jn++)
#pragma unroll
    for (int e = 0; e < 8; e++)
      out[(size_t)(q0 + e + hi * 8) * C_H + head * C_HD + jn * 16 + l16] =
          (bf16_t)(oacc[jn][e] * inv[e]);
}

// ---------------------------------------------------------------------------
// silu(g) * u -> bf16
// ---------------------------------------------------------------------------
__global__ void silumul_kernel(const float* __restrict__ g,
                               const float* __restrict__ u,
                               bf16_t* __restrict__ out, int n) {
  int idx = blockIdx.x * 256 + threadIdx.x;
  if (idx >= n) return;
  float gv = g[idx];
  float sg = gv / (1.0f + __expf(-gv));
  out[idx] = (bf16_t)(sg * u[idx]);
}

// ---------------------------------------------------------------------------
// Host-side launch
// ---------------------------------------------------------------------------
static inline void launch_gemm(const bf16_t* A, const float* B, float* C,
                               int M, int N, int K, int accumulate,
                               hipStream_t stream) {
  dim3 grid(N / 128, M / 256);
  gemm_bf16_wmma_kernel<<<grid, 256, 0, stream>>>(A, B, C, M, N, K, accumulate);
}

extern "C" void kernel_launch(void* const* d_in, const int* in_sizes, int n_in,
                              void* d_out, int out_size, void* d_ws, size_t ws_size,
                              hipStream_t stream) {
  const int*   ids      = (const int*)  d_in[0];
  const float* embed    = (const float*)d_in[1];
  const float* ln1      = (const float*)d_in[2];
  const float* Wq       = (const float*)d_in[3];
  const float* Wk       = (const float*)d_in[4];
  const float* Wv       = (const float*)d_in[5];
  const float* Wo       = (const float*)d_in[6];
  const float* ln2      = (const float*)d_in[7];
  const float* Wg       = (const float*)d_in[8];
  const float* Wu       = (const float*)d_in[9];
  const float* Wd       = (const float*)d_in[10];
  const float* final_ln = (const float*)d_in[11];
  float* logits = (float*)d_out;

  uintptr_t p = (uintptr_t)d_ws;
  auto carve = [&](size_t bytes) -> void* {
    void* r = (void*)p;
    p += (bytes + 255) & ~(size_t)255;
    return r;
  };
  float*  h     = (float*) carve((size_t)C_S * C_H * 4);
  float*  qbuf  = (float*) carve((size_t)C_S * C_H * 4);
  float*  kbuf  = (float*) carve((size_t)C_S * C_KV * C_HD * 4);
  float*  vbuf  = (float*) carve((size_t)C_S * C_KV * C_HD * 4);
  float*  gbuf  = (float*) carve((size_t)C_S * C_F * 4);
  float*  ubuf  = (float*) carve((size_t)C_S * C_F * 4);
  bf16_t* hnb   = (bf16_t*)carve((size_t)C_S * C_H * 2);
  bf16_t* attnb = (bf16_t*)carve((size_t)C_S * C_H * 2);
  bf16_t* gub   = (bf16_t*)carve((size_t)C_S * C_F * 2);
  bf16_t* qb16  = (bf16_t*)carve((size_t)C_S * C_H * 2);
  bf16_t* kb16  = (bf16_t*)carve((size_t)C_S * C_KV * C_HD * 2);
  bf16_t* vbt   = (bf16_t*)carve((size_t)C_KV * C_HD * C_S * 2);

  embed_gather_kernel<<<(C_S * C_H + 255) / 256, 256, 0, stream>>>(ids, embed, h);

  const int rope_q_total = C_S * C_NH * 32;
  const int rope_k_total = C_S * C_KV * 32;
  const int kv_n = C_S * C_KV * C_HD;
  const int ff_n = C_S * C_F;

  for (int i = 0; i < C_L; i++) {
    const float* wq = Wq + (size_t)i * C_H * C_H;
    const float* wk = Wk + (size_t)i * C_KV * C_HD * C_H;
    const float* wv = Wv + (size_t)i * C_KV * C_HD * C_H;
    const float* wo = Wo + (size_t)i * C_H * C_H;
    const float* wg = Wg + (size_t)i * C_F * C_H;
    const float* wu = Wu + (size_t)i * C_F * C_H;
    const float* wd = Wd + (size_t)i * C_H * C_F;

    rmsnorm_bf16_kernel<<<C_S, 256, 0, stream>>>(h, ln1 + (size_t)i * C_H, hnb);
    launch_gemm(hnb, wq, qbuf, C_S, C_H,         C_H, 0, stream);
    launch_gemm(hnb, wk, kbuf, C_S, C_KV * C_HD, C_H, 0, stream);
    launch_gemm(hnb, wv, vbuf, C_S, C_KV * C_HD, C_H, 0, stream);
    rope_bf16_kernel<<<(rope_q_total + 255) / 256, 256, 0, stream>>>(
        qbuf, qb16, C_NH, rope_q_total);
    rope_bf16_kernel<<<(rope_k_total + 255) / 256, 256, 0, stream>>>(
        kbuf, kb16, C_KV, rope_k_total);
    v_transpose_bf16_kernel<<<(kv_n + 255) / 256, 256, 0, stream>>>(vbuf, vbt);
    flash_attn_wmma_kernel<<<dim3(C_S / 128, C_NH), 256, 0, stream>>>(
        qb16, kb16, vbt, attnb);
    launch_gemm(attnb, wo, h, C_S, C_H, C_H, 1, stream);

    rmsnorm_bf16_kernel<<<C_S, 256, 0, stream>>>(h, ln2 + (size_t)i * C_H, hnb);
    launch_gemm(hnb, wg, gbuf, C_S, C_F, C_H, 0, stream);
    launch_gemm(hnb, wu, ubuf, C_S, C_F, C_H, 0, stream);
    silumul_kernel<<<(ff_n + 255) / 256, 256, 0, stream>>>(gbuf, ubuf, gub, ff_n);
    launch_gemm(gub, wd, h, C_S, C_H, C_F, 1, stream);
  }

  rmsnorm_bf16_kernel<<<C_S, 256, 0, stream>>>(h, final_ln, hnb);
  launch_gemm(hnb, embed, logits, C_S, C_V, C_H, 0, stream);
}
